// TimeLayer_62929860821278
// MI455X (gfx1250) — matrix operand order repaired and checked by reading, not compiled, not run on hardware
//
#include <hip/hip_runtime.h>
#include <cmath>

// ---------------------------------------------------------------------------
// Problem constants
// ---------------------------------------------------------------------------
#define BN    8        // batch
#define TN    2048     // time steps
#define DIN   896
#define DOUTN 128
#define RNK   16
#define HC    896      // lora_C hidden
#define NG    3584     // 4*HC
#define MROWS 16384    // B*T
#define NBLK  14       // recurrent workgroups (14 * 64 = 896 hidden units)

typedef _Float16 v8h  __attribute__((ext_vector_type(8)));
typedef _Float16 v16h __attribute__((ext_vector_type(16)));
typedef float    v8f  __attribute__((ext_vector_type(8)));
typedef int      v4i  __attribute__((ext_vector_type(4)));

#define CAT16(lo, hi) __builtin_shufflevector(lo, hi, 0,1,2,3,4,5,6,7,8,9,10,11,12,13,14,15)

// gfx1250 async global->LDS path (ASYNCcnt), if this toolchain declares it.
// Signature (from hipcc diagnostic): (v4i AS1* src, v4i AS3* dst, imm offset, imm cpol)
#if __has_builtin(__builtin_amdgcn_global_load_async_to_lds_b128) && \
    __has_builtin(__builtin_amdgcn_s_wait_asynccnt)
#define HAVE_ASYNC_LDS 1
typedef __attribute__((address_space(1))) v4i gas_v4i;
typedef __attribute__((address_space(3))) v4i las_v4i;
#else
#define HAVE_ASYNC_LDS 0
#endif

static __device__ __forceinline__ v8f wmma_f16(v16h a, v16h b, v8f c) {
    // D = A*B + C,  16x16x32, f16 in / f32 accumulate
    return __builtin_amdgcn_wmma_f32_16x16x32_f16(false, a, false, b, (short)0, c, false, false);
}

static __device__ __forceinline__ float sigmoidf_(float x) {
    return 1.0f / (1.0f + __expf(-x));
}

// ---------------------------------------------------------------------------
// K0: weight prep: f32 -> f16 conversions, K-padding, bias folding, state init
// ---------------------------------------------------------------------------
__global__ __launch_bounds__(256) void k_prep(
    const float* __restrict__ whhc,   // [3584,896]
    const float* __restrict__ wihd,   // [64,896]
    const float* __restrict__ whhd,   // [64,16]
    const float* __restrict__ Wb,     // [128,16]
    const float* __restrict__ bihc, const float* __restrict__ bhhc,
    const float* __restrict__ bihd, const float* __restrict__ bhhd,
    _Float16* __restrict__ whh16,     // [3584,896]
    _Float16* __restrict__ wihd16,    // [64,896]
    _Float16* __restrict__ whhd16,    // [64,32]  K padded 16->32 with zeros
    _Float16* __restrict__ wb16,      // [128,32] K padded 16->32 with zeros
    float* __restrict__ bc,           // [3584] = b_ih_c + b_hh_c
    float* __restrict__ bd,           // [64]   = b_ih_d + b_hh_d
    _Float16* __restrict__ hbuf,      // [2,16,896] zero
    unsigned* __restrict__ bar)
{
    size_t gid = (size_t)blockIdx.x * blockDim.x + threadIdx.x;
    size_t stride = (size_t)gridDim.x * blockDim.x;
    for (size_t i = gid; i < (size_t)NG * DIN; i += stride) whh16[i] = (_Float16)whhc[i];
    for (size_t i = gid; i < (size_t)64 * DIN; i += stride) wihd16[i] = (_Float16)wihd[i];
    for (size_t i = gid; i < 64 * 32; i += stride) {
        int n = (int)(i >> 5), k = (int)(i & 31);
        whhd16[i] = (k < 16) ? (_Float16)whhd[n * 16 + k] : (_Float16)0.f;
    }
    for (size_t i = gid; i < 128 * 32; i += stride) {
        int n = (int)(i >> 5), k = (int)(i & 31);
        wb16[i] = (k < 16) ? (_Float16)Wb[n * 16 + k] : (_Float16)0.f;
    }
    for (size_t i = gid; i < NG; i += stride) bc[i] = bihc[i] + bhhc[i];
    for (size_t i = gid; i < 64; i += stride) bd[i] = bihd[i] + bhhd[i];
    for (size_t i = gid; i < 2 * 16 * HC; i += stride) hbuf[i] = (_Float16)0.f;
    if (gid < 8) bar[gid] = 0u;
}

// ---------------------------------------------------------------------------
// K1: A = x @ Wa^T + ba     [8,2048,16]   (K=896 dot per element, VALU f32)
// ---------------------------------------------------------------------------
__global__ __launch_bounds__(256) void k_lora_a(
    const float* __restrict__ x, const float* __restrict__ Wa,
    const float* __restrict__ ba, float* __restrict__ Abuf)
{
    int o = blockIdx.x * 256 + threadIdx.x;     // 262144 outputs
    int row = o >> 4, j = o & 15;
    const float4* xr = (const float4*)(x + (size_t)row * DIN);
    const float4* wr = (const float4*)(Wa + (size_t)j * DIN);
    float s = ba[j];
    #pragma unroll 4
    for (int k = 0; k < DIN / 4; ++k) {
        float4 a = xr[k], b = wr[k];
        s += a.x * b.x + a.y * b.y + a.z * b.z + a.w * b.w;
    }
    Abuf[o] = s;
}

// ---------------------------------------------------------------------------
// K2: out = x @ Wo^T + bo + bb    WMMA GEMM  M=16384 N=128 K=896
//     grid = 1024 (M tiles), 8 waves/block -> each wave one 16-col N tile
// ---------------------------------------------------------------------------
__global__ __launch_bounds__(256) void k_orig(
    const float* __restrict__ x, const float* __restrict__ Wo,
    const float* __restrict__ bo, const float* __restrict__ bb,
    float* __restrict__ out)
{
    const int tid = threadIdx.x, l = tid & 31, w = tid >> 5;
    const int M0 = blockIdx.x * 16;
    const int n0 = w * 16;
    const int arow = l & 15, ah = (l >> 4) * 8;      // A-operand lane layout (16x32 f16)
    const int bcol = l & 15, bh = (l >> 4) * 8;      // B-operand lane layout (32x16 f16)
    const float* apb = x  + (size_t)(M0 + arow) * DIN + ah;
    const float* bpb = Wo + (size_t)(n0 + bcol) * DIN + bh;
    v8f acc = {};
    #pragma unroll 2
    for (int kk = 0; kk < 28; ++kk) {
        int kb = kk * 32;
        v16h a, b;
        #pragma unroll
        for (int e = 0; e < 8; ++e) { a[e] = (_Float16)apb[kb + e]; a[e + 8] = (_Float16)apb[kb + 16 + e]; }
        #pragma unroll
        for (int e = 0; e < 8; ++e) { b[e] = (_Float16)bpb[kb + e]; b[e + 8] = (_Float16)bpb[kb + 16 + e]; }
        acc = wmma_f16(a, b, acc);
    }
    float bias = bo[n0 + bcol] + bb[n0 + bcol];
    int mbase = (l >> 4) * 8;
    #pragma unroll
    for (int v = 0; v < 8; ++v)
        out[(size_t)(M0 + mbase + v) * DOUTN + n0 + bcol] = acc[v] + bias;
}

// ---------------------------------------------------------------------------
// K3: persistent cooperative recurrent kernel.
//  14 WGs x 256 threads. WG b owns hidden units [64b, 64b+64) of lora_C.
//  Per step: async-stage h into LDS -> WMMA gate GEMM -> activations ->
//  new h (f16) -> device barrier. WG0 additionally runs the lora_D LSTM step
//  and the Wb projection (WMMA) and accumulates into out.
// ---------------------------------------------------------------------------
__global__ __launch_bounds__(256, 1) void k_recurrent(
    const _Float16* __restrict__ whh16,   // [3584,896]
    const float*    __restrict__ wihc,    // [3584,16]
    const float*    __restrict__ bc,      // [3584]
    const _Float16* __restrict__ wihd16,  // [64,896]
    const _Float16* __restrict__ whhd16,  // [64,32]
    const float*    __restrict__ bd,      // [64]
    const _Float16* __restrict__ wb16,    // [128,32]
    const float*    __restrict__ Abuf,    // [8,2048,16]
    _Float16*       __restrict__ hbuf,    // [2,16,896] double buffered h
    unsigned*       __restrict__ bar,     // [0]=count [1]=phase
    float*          __restrict__ out)     // [8,2048,128]
{
    __shared__ _Float16 lds_h[16 * HC] __attribute__((aligned(16)));   // 28672 B
    __shared__ float    lds_gates[16 * 8 * 16];                        //  8192 B (rows 0..7 only)
    __shared__ float    lds_A[8 * 16];                                 //   512 B
    __shared__ float    lds_wihc[256 * 16];                            // 16384 B
    __shared__ float    lds_bias[256];                                 //  1024 B
    __shared__ _Float16 lds_hd[16 * 32] __attribute__((aligned(16)));  //  1024 B (K-padded h_d)
    __shared__ float    lds_gd[4 * 16 * 16];                           //  4096 B

    const int tid = threadIdx.x;
    const int l = tid & 31, w = tid >> 5;
    const int hbase = blockIdx.x * 64;
    const int arow = l & 15, ah = (l >> 4) * 8;
    const int bcol = l & 15, bh = (l >> 4) * 8;

    // Stage this WG's w_ih_c rows (gate-major: row = gate*64 + hj) + folded bias.
    {
        int gate = tid >> 6, hj = tid & 63;
        int gn = gate * HC + hbase + hj;
        #pragma unroll
        for (int k = 0; k < 16; ++k) lds_wihc[tid * 16 + k] = wihc[gn * 16 + k];
        lds_bias[tid] = bc[gn];
    }
    lds_hd[tid] = (_Float16)0.f;
    lds_hd[tid + 256] = (_Float16)0.f;

    float c0 = 0.f, c1 = 0.f;   // lora_C cell state, 2 elements per thread
    float cd = 0.f;             // lora_D cell state (threads < 128 of WG0)

    for (int t = 0; t < TN; ++t) {
        // ------- stage h_{t-1} and A_t into LDS -------
        {
#if HAVE_ASYNC_LDS
            // 28 KB h tile: async global->LDS (ASYNCcnt), 7 x b128 per thread.
            char* srcg = (char*)(hbuf + (size_t)(t & 1) * 16 * HC);
            char* dstl = (char*)lds_h;
            #pragma unroll
            for (int i = 0; i < 7; ++i) {
                int off = (i * 256 + tid) * 16;
                __builtin_amdgcn_global_load_async_to_lds_b128(
                    (gas_v4i*)(srcg + off), (las_v4i*)(dstl + off), 0, 0);
            }
#else
            const uint4* src = (const uint4*)(hbuf + (size_t)(t & 1) * 16 * HC);
            uint4* dst = (uint4*)lds_h;
            #pragma unroll
            for (int i = 0; i < 7; ++i) dst[i * 256 + tid] = src[i * 256 + tid];
#endif
            if (tid < 128) {
                int r = tid >> 4, j = tid & 15;
                lds_A[tid] = Abuf[((size_t)r * TN + t) * RNK + j];
            }
#if HAVE_ASYNC_LDS
            __builtin_amdgcn_s_wait_asynccnt(0);
#endif
        }
        __syncthreads();

        // ------- gate GEMM: 2 gate tiles per wave, K=896 (28 WMMA ksteps) -------
        #pragma unroll
        for (int s = 0; s < 2; ++s) {
            const int p = w * 2 + s;                 // gate tile id 0..15
            const int gate = p >> 2, ht = p & 3;
            const int n0 = gate * HC + hbase + ht * 16;
            const _Float16* bpb = whh16 + (size_t)(n0 + bcol) * DIN + bh;
            v8f acc = {};
            #pragma unroll 4
            for (int kk = 0; kk < 28; ++kk) {
                const int kb = kk * 32;
                const _Float16* ap = &lds_h[arow * HC + kb + ah];
                v8h alo = *(const v8h*)ap;
                v8h ahi = *(const v8h*)(ap + 16);
                const _Float16* bp = bpb + kb;
                v8h blo = *(const v8h*)bp;
                v8h bhi = *(const v8h*)(bp + 16);
                acc = wmma_f16(CAT16(alo, ahi), CAT16(blo, bhi), acc);
            }
            if (l < 16) {                            // rows 0..7 (real batches) only
                #pragma unroll
                for (int v = 0; v < 8; ++v)
                    lds_gates[(p * 8 + v) * 16 + bcol] = acc[v];
            }
        }
        __syncthreads();

        // ------- activations + cell update (512 live values, 2/thread) -------
        #pragma unroll
        for (int s2 = 0; s2 < 2; ++s2) {
            const int v = tid + s2 * 256;
            const int hj = v >> 3, r = v & 7;
            const int ht = hj >> 4, col = hj & 15;
            float g4[4];
            #pragma unroll
            for (int g = 0; g < 4; ++g) {
                float a = lds_bias[g * 64 + hj];
                const float* wr = &lds_wihc[(g * 64 + hj) * 16];
                const float* ar = &lds_A[r * 16];
                #pragma unroll
                for (int k = 0; k < 16; ++k) a += wr[k] * ar[k];   // xg on the fly
                g4[g] = a + lds_gates[((g * 4 + ht) * 8 + r) * 16 + col];
            }
            const float ii = sigmoidf_(g4[0]);
            const float ff = sigmoidf_(g4[1]);
            const float gg = tanhf(g4[2]);
            const float oo = sigmoidf_(g4[3]);
            const float cold = s2 ? c1 : c0;
            const float cnew = ff * cold + ii * gg;
            if (s2) c1 = cnew; else c0 = cnew;
            const float hh = oo * tanhf(cnew);
            hbuf[(size_t)((t + 1) & 1) * 16 * HC + r * HC + hbase + hj] = (_Float16)hh;
        }
        __threadfence();                              // release h_t to device scope

        // ------- device-wide step barrier -------
        __builtin_amdgcn_s_cluster_barrier();         // CDNA5 cluster barrier (S_NOP if not clustered)
        __syncthreads();
        if (tid == 0) {
            unsigned old = __hip_atomic_fetch_add(&bar[0], 1u, __ATOMIC_ACQ_REL, __HIP_MEMORY_SCOPE_AGENT);
            if (old == NBLK - 1u) {
                __hip_atomic_store(&bar[0], 0u, __ATOMIC_RELAXED, __HIP_MEMORY_SCOPE_AGENT);
                __hip_atomic_fetch_add(&bar[1], 1u, __ATOMIC_ACQ_REL, __HIP_MEMORY_SCOPE_AGENT);
            } else {
                while (__hip_atomic_load(&bar[1], __ATOMIC_ACQUIRE, __HIP_MEMORY_SCOPE_AGENT) < (unsigned)(t + 1))
                    __builtin_amdgcn_s_sleep(2);
            }
        }
        __syncthreads();

        // ------- WG0: lora_D LSTM step + Wb projection for step t -------
        if (blockIdx.x == 0) {
            const _Float16* ct = hbuf + (size_t)((t + 1) & 1) * 16 * HC;   // C_t
            if (w < 4) {                              // 4 gate tiles of g_d (N=64)
                const int n0 = w * 16;
                const _Float16* bpb = wihd16 + (size_t)(n0 + bcol) * DIN + bh;
                v8f acc = {};
                #pragma unroll 4
                for (int kk = 0; kk < 28; ++kk) {
                    const int kb = kk * 32;
                    const _Float16* ap = ct + arow * HC + kb + ah;
                    v8h alo = *(const v8h*)ap;
                    v8h ahi = *(const v8h*)(ap + 16);
                    const _Float16* bp = bpb + kb;
                    v8h blo = *(const v8h*)bp;
                    v8h bhi = *(const v8h*)(bp + 16);
                    acc = wmma_f16(CAT16(alo, ahi), CAT16(blo, bhi), acc);
                }
                // recurrent term: h_d (LDS, K padded to 32) x w_hh_d^T
                {
                    const _Float16* ap = &lds_hd[arow * 32 + ah];
                    v8h alo = *(const v8h*)ap;
                    v8h ahi = *(const v8h*)(ap + 16);
                    const _Float16* bp = whhd16 + (size_t)(n0 + bcol) * 32 + bh;
                    v8h blo = *(const v8h*)bp;
                    v8h bhi = *(const v8h*)(bp + 16);
                    acc = wmma_f16(CAT16(alo, ahi), CAT16(blo, bhi), acc);
                }
                const int mbase = (l >> 4) * 8;
                #pragma unroll
                for (int v = 0; v < 8; ++v)
                    lds_gd[(w * 16 + mbase + v) * 16 + bcol] = acc[v];
            }
            __syncthreads();
            if (tid < 128) {                          // D activations: 8 rows x 16 hid
                const int dj = tid >> 3, r = tid & 7;
                const float gi = lds_gd[(0 * 16 + r) * 16 + dj] + bd[dj];
                const float gf = lds_gd[(1 * 16 + r) * 16 + dj] + bd[16 + dj];
                const float gg = lds_gd[(2 * 16 + r) * 16 + dj] + bd[32 + dj];
                const float go = lds_gd[(3 * 16 + r) * 16 + dj] + bd[48 + dj];
                cd = sigmoidf_(gf) * cd + sigmoidf_(gi) * tanhf(gg);
                lds_hd[r * 32 + dj] = (_Float16)(sigmoidf_(go) * tanhf(cd));
            }
            __syncthreads();
            {                                         // Bout = h_d @ Wb^T, wave w: cols [16w,16w+16)
                const int n0 = w * 16;
                const _Float16* ap = &lds_hd[arow * 32 + ah];
                v8h alo = *(const v8h*)ap;
                v8h ahi = *(const v8h*)(ap + 16);
                const _Float16* bp = wb16 + (size_t)(n0 + bcol) * 32 + bh;
                v8h blo = *(const v8h*)bp;
                v8h bhi = *(const v8h*)(bp + 16);
                v8f acc = {};
                acc = wmma_f16(CAT16(alo, ahi), CAT16(blo, bhi), acc);
                if (l < 16) {
                    #pragma unroll
                    for (int v = 0; v < 8; ++v) {
                        size_t o = ((size_t)v * TN + t) * DOUTN + n0 + bcol;
                        out[o] += acc[v];
                    }
                }
            }
            __syncthreads();
        }
    }
}

// ---------------------------------------------------------------------------
// Host launcher
// ---------------------------------------------------------------------------
extern "C" void kernel_launch(void* const* d_in, const int* in_sizes, int n_in,
                              void* d_out, int out_size, void* d_ws, size_t ws_size,
                              hipStream_t stream) {
    (void)in_sizes; (void)n_in; (void)out_size; (void)ws_size;
    const float* x     = (const float*)d_in[0];
    const float* Wo    = (const float*)d_in[1];
    const float* bo    = (const float*)d_in[2];
    const float* Wa    = (const float*)d_in[3];
    const float* ba    = (const float*)d_in[4];
    const float* wihc  = (const float*)d_in[5];
    const float* whhc  = (const float*)d_in[6];
    const float* bihc  = (const float*)d_in[7];
    const float* bhhc  = (const float*)d_in[8];
    const float* wihd  = (const float*)d_in[9];
    const float* whhd  = (const float*)d_in[10];
    const float* bihd  = (const float*)d_in[11];
    const float* bhhd  = (const float*)d_in[12];
    const float* Wb    = (const float*)d_in[13];
    const float* bb    = (const float*)d_in[14];
    float* out = (float*)d_out;

    char* ws = (char*)d_ws;
    _Float16* whh16  = (_Float16*)(ws + 0);            // 6,422,528 B
    _Float16* wihd16 = (_Float16*)(ws + 6422528);      //   114,688 B
    _Float16* whhd16 = (_Float16*)(ws + 6537216);      //     4,096 B
    _Float16* wb16   = (_Float16*)(ws + 6541312);      //     8,192 B
    float*    bc     = (float*)   (ws + 6549504);      //    14,336 B
    float*    bd     = (float*)   (ws + 6563840);      //       256 B
    _Float16* hbuf   = (_Float16*)(ws + 6564096);      //    57,344 B
    unsigned* bar    = (unsigned*)(ws + 6621440);      //       256 B
    float*    Abuf   = (float*)   (ws + 6621696);      // 1,048,576 B  (total ~7.3 MB)

    k_prep<<<4096, 256, 0, stream>>>(whhc, wihd, whhd, Wb, bihc, bhhc, bihd, bhhd,
                                     whh16, wihd16, whhd16, wb16, bc, bd, hbuf, bar);
    k_lora_a<<<1024, 256, 0, stream>>>(x, Wa, ba, Abuf);
    k_orig<<<1024, 256, 0, stream>>>(x, Wo, bo, bb, out);
    k_recurrent<<<NBLK, 256, 0, stream>>>(whh16, wihc, bc, wihd16, whhd16, bd, wb16,
                                          Abuf, hbuf, bar, out);
}
